// CrossAttention1D_78640851190158
// MI455X (gfx1250) — compile-verified, matrix operand order checked by reading, not run on hardware
//
#include <hip/hip_runtime.h>
#include <hip/hip_bf16.h>

typedef __attribute__((ext_vector_type(16))) _Float16 v16h;
typedef __attribute__((ext_vector_type(8)))  _Float16 v8h;
typedef __attribute__((ext_vector_type(8)))  float    v8f;
typedef __attribute__((ext_vector_type(4)))  float    f4v;

#define CDIM   512
#define CONDD  256
#define BATCH  8
#define TLEN   1024
#define LPADH  (CDIM + 8)   // f16 LDS row stride: 520 halfs = 1040B (16B aligned),
                            // 260 dwords % 64 banks = 4 -> conflict-free column access

// -----------------------------------------------------------------------------
// Stage 1 (degenerate attention collapsed): for the 8 batch rows,
//   r = proj(out(wv2(Wv @ cond + bv) + bv2) + out_b) + proj_b
// All 8 rows fit in ONE 16-row WMMA tile (rows 8..15 zero). 32 wave32s, each
// owning one 16-wide N tile (32*16 = 512 = C). Inter-stage activations live in
// LDS as f16 (exactly the precision WMMA consumes) -> A fragments are two
// ds_load_b128 per WMMA. Weights stream from global with pk-cvt to f16;
// next-stage weight rows are global_prefetch'ed ahead of time.
// -----------------------------------------------------------------------------
__global__ __launch_bounds__(1024)
void ca1d_stage1_wmma(const float* __restrict__ cond,
                      const float* __restrict__ Wv,        const float* __restrict__ bv,
                      const float* __restrict__ in_proj_w, const float* __restrict__ in_proj_b,
                      const float* __restrict__ out_w,     const float* __restrict__ out_b,
                      const float* __restrict__ proj_w,    const float* __restrict__ proj_b,
                      float* __restrict__ r_out)
{
    __shared__ __align__(16) _Float16 bufA[16][LPADH];
    __shared__ __align__(16) _Float16 bufB[16][LPADH];

    const int tid  = threadIdx.x;
    const int lane = tid & 31;
    const int wave = tid >> 5;      // 0..31 -> N tile index
    const int lm   = lane & 15;     // M (A) / N-in-tile (B, C, D)
    const int hi   = lane >> 4;     // lane-half selector
    const int koff = hi * 8;        // K offset of this half's fragment slice

    const float* Wmat[4] = { Wv, in_proj_w + 2 * CDIM * CDIM, out_w, proj_w };
    const float* Bias[4] = { bv, in_proj_b + 2 * CDIM,        out_b, proj_b };
    const int    Kdim[4] = { CONDD, CDIM, CDIM, CDIM };

    const int n = wave * 16 + lm;   // this lane's global output column

    // Warm stage-0 weight rows into cache while we populate LDS.
    {
        const float* w0 = Wv + (size_t)n * CONDD;       // 1 KB row
        #pragma unroll
        for (int pb = 0; pb < CONDD * 4; pb += 128)
            __builtin_prefetch((const char*)w0 + pb, 0, 0);
    }

    // Init activations: rows 0..7 = cond (cols 0..255), everything else zero.
    for (int i = tid; i < 16 * CDIM; i += 1024) {
        const int m = i / CDIM, k = i % CDIM;
        float v = 0.0f;
        if (m < BATCH && k < CONDD) v = cond[m * CONDD + k];
        bufA[m][k] = (_Float16)v;
    }
    __syncthreads();

    _Float16 (*src)[LPADH] = bufA;
    _Float16 (*dst)[LPADH] = bufB;

    #pragma unroll
    for (int s = 0; s < 4; ++s) {
        const float* wrow = Wmat[s] + (size_t)n * Kdim[s];  // B[k][n] = W[n][k]
        const float  bn   = Bias[s][n];
        const int    K    = Kdim[s];

        // Overlap next stage's weight fetch with this stage's compute.
        if (s < 3) {
            const float* nw = Wmat[s + 1] + (size_t)n * Kdim[s + 1];   // 2 KB row
            #pragma unroll
            for (int pb = 0; pb < CDIM * 4; pb += 128)
                __builtin_prefetch((const char*)nw + pb, 0, 0);
        }

        v8f acc = {};
        for (int k0 = 0; k0 < K; k0 += 32) {
            // A 16x32 f16 fragment: lane lm = row M; this half covers
            // K = k0+koff..+7 and k0+16+koff..+7  -> two 16B LDS loads.
            v8h alo = *(const v8h*)&src[lm][k0 + koff];
            v8h ahi = *(const v8h*)&src[lm][k0 + 16 + koff];
            v16h a = __builtin_shufflevector(alo, ahi,
                         0, 1, 2, 3, 4, 5, 6, 7, 8, 9, 10, 11, 12, 13, 14, 15);
            // B 32x16 f16 fragment: lane lm = col N; contiguous K reads of W row.
            v16h b;
            #pragma unroll
            for (int j = 0; j < 8; ++j) {
                b[j]     = (_Float16)wrow[k0 + koff + j];
                b[j + 8] = (_Float16)wrow[k0 + 16 + koff + j];
            }
            acc = __builtin_amdgcn_wmma_f32_16x16x32_f16(
                      false, a, false, b, (short)0, acc, false, false);
        }

        if (s < 3) {
            // C/D f32 layout: VGPR rr -> row (hi*8 + rr), lane lm -> col n.
            #pragma unroll
            for (int rr = 0; rr < 8; ++rr)
                dst[hi * 8 + rr][n] = (_Float16)(acc[rr] + bn);
            __syncthreads();
            _Float16 (*tmp)[LPADH] = src; src = dst; dst = tmp;
        } else {
            // Final stage: lanes with hi==0 hold exactly batch rows 0..7 -> f32 out.
            if (hi == 0) {
                #pragma unroll
                for (int rr = 0; rr < 8; ++rr)
                    r_out[rr * CDIM + n] = acc[rr] + bn;
            }
        }
    }
}

// -----------------------------------------------------------------------------
// Stage 2: out[b,c,t] = x[b,c,t] + r[b,c]. One 256-thread block per (b,c) row;
// r[row] is block-uniform (scalar load). Non-temporal b128 streaming:
// 33.6 MB @ 23.3 TB/s ~= 1.4 us, pure HBM roofline.
// -----------------------------------------------------------------------------
__global__ __launch_bounds__(256)
void ca1d_bcast_add(const float* __restrict__ x,
                    const float* __restrict__ r,
                    float* __restrict__ out)
{
    const int row = blockIdx.x;                 // row = b*CDIM + c, 0..4095
    const float rv = r[row];
    const f4v* __restrict__ xi = (const f4v*)(x   + (size_t)row * TLEN);
    f4v* __restrict__       oi = (f4v*)      (out + (size_t)row * TLEN);
    f4v v = __builtin_nontemporal_load(xi + threadIdx.x);
    v.x += rv; v.y += rv; v.z += rv; v.w += rv;
    __builtin_nontemporal_store(v, oi + threadIdx.x);
}

extern "C" void kernel_launch(void* const* d_in, const int* in_sizes, int n_in,
                              void* d_out, int out_size, void* d_ws, size_t ws_size,
                              hipStream_t stream)
{
    (void)in_sizes; (void)n_in; (void)out_size; (void)ws_size;
    const float* x         = (const float*)d_in[0];
    const float* cond      = (const float*)d_in[1];
    // d_in[2..7]: ln_g, ln_b, Wq, bq, Wk, bk — mathematically unused:
    // scores are constant along s => softmax is exactly uniform => attn out = V.
    const float* Wv        = (const float*)d_in[8];
    const float* bv        = (const float*)d_in[9];
    const float* in_proj_w = (const float*)d_in[10];
    const float* in_proj_b = (const float*)d_in[11];
    const float* out_w     = (const float*)d_in[12];
    const float* out_b     = (const float*)d_in[13];
    const float* proj_w    = (const float*)d_in[14];
    const float* proj_b    = (const float*)d_in[15];
    float* out = (float*)d_out;
    float* r   = (float*)d_ws;   // 8*512 floats = 16 KB scratch

    hipLaunchKernelGGL(ca1d_stage1_wmma, dim3(1), dim3(1024), 0, stream,
                       cond, Wv, bv, in_proj_w, in_proj_b,
                       out_w, out_b, proj_w, proj_b, r);
    hipLaunchKernelGGL(ca1d_bcast_add, dim3(BATCH * CDIM), dim3(256), 0, stream,
                       x, r, out);
}
